// LSTMModel_49117245997215
// MI455X (gfx1250) — matrix operand order repaired and checked by reading, not compiled, not run on hardware
//
#include <hip/hip_runtime.h>
#include <cstddef>

typedef __attribute__((ext_vector_type(16))) _Float16 v16h;
typedef __attribute__((ext_vector_type(8)))  _Float16 v8h;
typedef __attribute__((ext_vector_type(8)))  float    v8f;

#define T_STEPS 500
#define IN_DIM  40
#define H_DIM   64
// Combined K layout (f16 in LDS): [0..63] = h, [64..103] = x, [104..127] = zero pad
#define K_COMB  104
#define K_PAD   128

__device__ __forceinline__ float fast_tanh(float z) {
    float e = __builtin_amdgcn_exp2f(z * -2.8853900817779268f);
    return __builtin_fmaf(__builtin_amdgcn_rcpf(1.0f + e), 2.0f, -1.0f);
}

__launch_bounds__(128, 1)
__global__ void lstm_fused_kernel(const float* __restrict__ X,
                                  const float* __restrict__ W_ih,
                                  const float* __restrict__ W_hh,
                                  const float* __restrict__ b_ih,
                                  const float* __restrict__ b_hh,
                                  const float* __restrict__ fc1_w,
                                  const float* __restrict__ fc1_b,
                                  const float* __restrict__ fc2_w,
                                  const float* __restrict__ fc2_b,
                                  float* __restrict__ out)
{
    __shared__ _Float16 sAh[16][K_PAD];   // combined activation (f16): h | x | 0-pad
    __shared__ float    sG[4][16][H_DIM]; // activated gates i,f,g,o

    const int tid   = threadIdx.x;        // 0..127
    const int lane  = tid & 31;
    const int wv    = tid >> 5;           // wave id == gate id (i,f,g,o)
    const int col   = lane & 15;
    const int hi    = lane >> 4;          // 0 / 1 half-wave
    const int khalf = hi * 16;            // B-fragment K sub-base
    const int btile = blockIdx.x;
    const bool is_g = (__builtin_amdgcn_readfirstlane(wv) == 2);

    // Wave-uniform branch-free activation params: sigmoid (i,f,o) / tanh (g)
    const float zscale = is_g ? -2.8853900817779268f : -1.4426950408889634f;
    const float omul   = is_g ?  2.0f : 1.0f;
    const float oadd   = is_g ? -1.0f : 0.0f;

    // ---- Preload B fragments (combined weights -> f16) + bias, VGPR-resident ----
    v16h Bf[4][4];
    float bzs[4];   // bias * zscale, folds into exp2 argument via FMA
#pragma unroll
    for (int nt = 0; nt < 4; ++nt) {
        const int n = wv * 64 + nt * 16 + col;
        bzs[nt] = (b_ih[n] + b_hh[n]) * zscale;
#pragma unroll
        for (int kc = 0; kc < 4; ++kc) {
            v16h b;
#pragma unroll
            for (int e = 0; e < 16; ++e) {
                const int k = kc * 32 + khalf + e;
                float v = 0.f;
                if (k < 64)           v = W_hh[n * H_DIM + k];
                else if (k < K_COMB)  v = W_ih[n * IN_DIM + (k - 64)];
                b[e] = (_Float16)v;
            }
            Bf[nt][kc] = b;
        }
    }

    // ---- Init LDS (h + x + pad = 0; pad stays 0 forever) ----
    for (int idx = tid; idx < 16 * K_PAD; idx += 128) {
        sAh[idx >> 7][idx & 127] = (_Float16)0.f;
    }
    // cell state in registers: thread owns row m = tid>>3, cols (tid&7)*8..+8
    float creg[8];
#pragma unroll
    for (int q = 0; q < 8; ++q) creg[q] = 0.f;

    // ---- Per-thread x staging geometry: 40 = 8 * 5 factors exactly ----
    const int xm = tid >> 3;              // row 0..15
    const int xk = (tid & 7) * 5;         // col 0,5,..,35 (5 contiguous floats)
    const float* xptr = X + (((size_t)(btile * 16 + xm)) * T_STEPS + 1) * IN_DIM + xk;
    _Float16*    xsp  = &sAh[xm][64 + xk];

    // Stage x[t=0]
    {
        const float* p0 = xptr - IN_DIM;
#pragma unroll
        for (int q = 0; q < 5; ++q) xsp[q] = (_Float16)p0[q];
    }
    __syncthreads();

    // ---- Recurrent loop ----
    for (int t = 0; t < T_STEPS; ++t) {
        // Build A fragments: two aligned 16B LDS reads per fragment
        v16h Af[4];
        {
            const v8h* rp = (const v8h*)&sAh[col][0];
#pragma unroll
            for (int kc = 0; kc < 4; ++kc) {
                const int c0 = kc * 4 + hi * 2;
                v8h lo = rp[c0];
                v8h hh = rp[c0 + 1];
                Af[kc] = __builtin_shufflevector(lo, hh,
                           0,1,2,3,4,5,6,7,8,9,10,11,12,13,14,15);
            }
        }
        __syncthreads();   // frag reads done before x/h regions are rewritten

        // Stage next x tile: batched loads, one wait, contiguous DS stores
        if (t + 1 < T_STEPS) {
            float xv[5];
#pragma unroll
            for (int q = 0; q < 5; ++q) xv[q] = xptr[q];
            if (t + 2 < T_STEPS)
                __builtin_prefetch(xptr + IN_DIM, 0, 1);
#pragma unroll
            for (int q = 0; q < 5; ++q) xsp[q] = (_Float16)xv[q];
            xptr += IN_DIM;
        }

        // z = a @ Wcomb^T : 16 WMMAs per wave, accumulators start at 0
        v8f acc[4];
#pragma unroll
        for (int nt = 0; nt < 4; ++nt) {
            v8f c = { 0.f, 0.f, 0.f, 0.f, 0.f, 0.f, 0.f, 0.f };
#pragma unroll
            for (int kc = 0; kc < 4; ++kc) {
                c = __builtin_amdgcn_wmma_f32_16x16x32_f16(
                        false, Af[kc], false, Bf[nt][kc], (short)0, c, false, false);
            }
            acc[nt] = c;
        }

        // Branch-free per-wave activation: a = omul*rcp(1+exp2(z*zscale+b*zscale))+oadd
        const int rbase = hi * 8;
#pragma unroll
        for (int nt = 0; nt < 4; ++nt) {
#pragma unroll
            for (int r = 0; r < 8; ++r) {
                const float e = __builtin_amdgcn_exp2f(
                                    __builtin_fmaf(acc[nt][r], zscale, bzs[nt]));
                const float s = __builtin_amdgcn_rcpf(1.0f + e);
                sG[wv][rbase + r][nt * 16 + col] = __builtin_fmaf(s, omul, oadd);
            }
        }
        __syncthreads();

        // Cooperative c/h update: c = f*c + i*g ; h = o*tanh(c)  (c in VGPRs)
        {
            const int m  = tid >> 3;
            const int c0 = (tid & 7) * 8;
#pragma unroll
            for (int q = 0; q < 8; ++q) {
                const int k = c0 + q;
                const float gi = sG[0][m][k];
                const float gf = sG[1][m][k];
                const float gg = sG[2][m][k];
                const float go = sG[3][m][k];
                const float cc = __builtin_fmaf(gf, creg[q], gi * gg);
                creg[q] = cc;
                sAh[m][k] = (_Float16)(go * fast_tanh(cc));
            }
        }
        __syncthreads();
    }

    // ---- fc1: relu(h @ fc1_w^T + b) -> reuse sG[0] as [16][64] ----
    {
        const int m  = tid >> 3;
        const int j0 = (tid & 7) * 8;
#pragma unroll
        for (int q = 0; q < 8; ++q) {
            const int j = j0 + q;
            float acc = fc1_b[j];
            for (int k = 0; k < H_DIM; ++k)
                acc += (float)sAh[m][k] * fc1_w[j * H_DIM + k];
            sG[0][m][j] = acc > 0.f ? acc : 0.f;
        }
    }
    __syncthreads();

    // ---- fc2: [16,64] @ [64,2]^T + b -> d_out ----
    if (tid < 32) {
        const int m = tid >> 1, j = tid & 1;
        float acc = fc2_b[j];
        for (int k = 0; k < H_DIM; ++k)
            acc += sG[0][m][k] * fc2_w[j * H_DIM + k];
        out[(size_t)(btile * 16 + m) * 2 + j] = acc;
    }
}

extern "C" void kernel_launch(void* const* d_in, const int* in_sizes, int n_in,
                              void* d_out, int out_size, void* d_ws, size_t ws_size,
                              hipStream_t stream)
{
    (void)n_in; (void)out_size; (void)d_ws; (void)ws_size;
    const float* X     = (const float*)d_in[0];
    const float* W_ih  = (const float*)d_in[1];
    const float* W_hh  = (const float*)d_in[2];
    const float* b_ih  = (const float*)d_in[3];
    const float* b_hh  = (const float*)d_in[4];
    const float* fc1_w = (const float*)d_in[5];
    const float* fc1_b = (const float*)d_in[6];
    const float* fc2_w = (const float*)d_in[7];
    const float* fc2_b = (const float*)d_in[8];
    float* out = (float*)d_out;

    const int B = in_sizes[0] / (T_STEPS * IN_DIM);  // 1024
    const int grid = B / 16;                          // 64 workgroups x 16 rows

    lstm_fused_kernel<<<grid, 128, 0, stream>>>(X, W_ih, W_hh, b_ih, b_hh,
                                                fc1_w, fc1_b, fc2_w, fc2_b, out);
}
